// ShiftWindowsTransformerBlock_21449066676782
// MI455X (gfx1250) — matrix operand order repaired and checked by reading, not compiled, not run on hardware
//
#include <hip/hip_runtime.h>
#include <hip/hip_bf16.h>

// ---------------------------------------------------------------------------
// Swin shifted-window transformer block for MI455X (gfx1250, wave32, WMMA).
// All GEMMs use v_wmma_f32_16x16x32_f16 (f16 in / f32 accumulate).
// ---------------------------------------------------------------------------

typedef __attribute__((ext_vector_type(16))) _Float16 v16h;
typedef __attribute__((ext_vector_type(8)))  _Float16 v8h;
typedef __attribute__((ext_vector_type(8)))  float    v8f;

#define HIMG   128
#define WSZ    8
#define SHIFT_ 4
#define NHEAD  3
#define CDIM   96
#define NTOK   64                 // tokens per window
#define NWIN   256                // windows per image
#define BATCH  16
#define MROWS  (BATCH * NWIN * NTOK)   // 262144 window-tokens

// ---------------- WMMA fragment helpers (contiguous-k operand order, wave32) -------

__device__ __forceinline__ v16h make_frag(v8h lo, v8h hi) {
  v16h r;
#pragma unroll
  for (int e = 0; e < 8; ++e) { r[e] = lo[e]; r[e + 8] = hi[e]; }
  return r;
}

// A matrix 16x32 f16, row-major source with leading dim ldA (halfs).
// lane L holds row m0+(L&15); K = kb + 16*g + h (contiguous 16 halfs), g=L>>4.
__device__ __forceinline__ v16h load_a_frag(const _Float16* A, int ldA,
                                            int m0, int kb, int lane) {
  const int row = m0 + (lane & 15);
  const int g   = lane >> 4;
  const _Float16* p = A + (size_t)row * ldA + kb + 16 * g;
  v8h lo = *reinterpret_cast<const v8h*>(p);
  v8h hi = *reinterpret_cast<const v8h*>(p + 8);
  return make_frag(lo, hi);
}

// B matrix 32x16 f16 supplied as BT (NxK row-major, leading dim ldB halfs):
// lane L holds col n0+(L&15); K = kb + 16*(L>>4) + h  (contiguous 16 halfs).
__device__ __forceinline__ v16h load_b_frag(const _Float16* BT, int ldB,
                                            int n0, int kb, int lane) {
  const int col = n0 + (lane & 15);
  const int g   = lane >> 4;
  const _Float16* p = BT + (size_t)col * ldB + kb + 16 * g;
  v8h lo = *reinterpret_cast<const v8h*>(p);
  v8h hi = *reinterpret_cast<const v8h*>(p + 8);
  return make_frag(lo, hi);
}

__device__ __forceinline__ v8f wmma_f16(v16h a, v16h b, v8f c) {
  return __builtin_amdgcn_wmma_f32_16x16x32_f16(false, a, false, b,
                                                (short)0, c, false, false);
}

__device__ __forceinline__ v8f vzero8() {
  v8f z = {0.f, 0.f, 0.f, 0.f, 0.f, 0.f, 0.f, 0.f};
  return z;
}

// ---------------- small prep kernels ---------------------------------------

// W (KxN f32, row-major) -> WT (NxK f16, row-major) so B-fragments are
// contiguous 32-byte loads.
__global__ void k_weight_to_bt(const float* __restrict__ W,
                               _Float16* __restrict__ WT, int K, int N) {
  int idx = blockIdx.x * blockDim.x + threadIdx.x;
  if (idx >= K * N) return;
  int k = idx / N, n = idx % N;
  WT[(size_t)n * K + k] = (_Float16)W[idx];
}

// relative-position bias gather: biasHNN[head][i][j] = table[relidx(i,j)][head]
__global__ void k_bias_gather(const float* __restrict__ table,
                              float* __restrict__ biasHNN) {
  int idx = blockIdx.x * blockDim.x + threadIdx.x;  // NHEAD*64*64
  if (idx >= NHEAD * NTOK * NTOK) return;
  int head = idx / (NTOK * NTOK);
  int ij   = idx % (NTOK * NTOK);
  int i = ij >> 6, j = ij & 63;
  int dh = (i >> 3) - (j >> 3) + (WSZ - 1);
  int dw = (i & 7)  - (j & 7)  + (WSZ - 1);
  biasHNN[idx] = table[(dh * (2 * WSZ - 1) + dw) * NHEAD + head];
}

// ---------------- LayerNorm (one wave per token, C=96 -> 3 vals/lane) ------
// windowed=1: read source pixel via cyclic shift + window partition, write
//             window-token-major f16 rows (for the QKV GEMM A-matrix).
// windowed=0: identity token mapping (LN2 of x1).
__global__ __launch_bounds__(256)
void k_layernorm(const float* __restrict__ x, const float* __restrict__ gvec,
                 const float* __restrict__ bvec, _Float16* __restrict__ out,
                 int windowed) {
  const int wid  = threadIdx.x >> 5;
  const int lane = threadIdx.x & 31;
  const int m    = blockIdx.x * 8 + wid;   // destination token 0..MROWS-1

  size_t src_row;
  if (windowed) {
    int wg = m >> 6, t = m & 63;
    int b = wg >> 8, wimg = wg & 255;
    int wr = wimg >> 4, wc = wimg & 15;
    int hh = wr * WSZ + (t >> 3), ww = wc * WSZ + (t & 7);
    int h0 = (hh + SHIFT_) & (HIMG - 1);
    int w0 = (ww + SHIFT_) & (HIMG - 1);
    src_row = ((size_t)b * (HIMG * HIMG) + h0 * HIMG + w0) * CDIM;
  } else {
    src_row = (size_t)m * CDIM;
  }

  const float* row = x + src_row;
  float v0 = row[lane], v1 = row[lane + 32], v2 = row[lane + 64];
  float s = v0 + v1 + v2;
#pragma unroll
  for (int msk = 1; msk < 32; msk <<= 1) s += __shfl_xor(s, msk, 32);
  float mean = s * (1.f / 96.f);
  float d0 = v0 - mean, d1 = v1 - mean, d2 = v2 - mean;
  float sq = d0 * d0 + d1 * d1 + d2 * d2;
#pragma unroll
  for (int msk = 1; msk < 32; msk <<= 1) sq += __shfl_xor(sq, msk, 32);
  float rstd = rsqrtf(sq * (1.f / 96.f) + 1e-5f);

  _Float16* orow = out + (size_t)m * CDIM;
  orow[lane]      = (_Float16)(d0 * rstd * gvec[lane]      + bvec[lane]);
  orow[lane + 32] = (_Float16)(d1 * rstd * gvec[lane + 32] + bvec[lane + 32]);
  orow[lane + 64] = (_Float16)(d2 * rstd * gvec[lane + 64] + bvec[lane + 64]);
}

// ---------------- QKV GEMM: (262144x96) x (96x288) -------------------------
// Each wave: one 16-row M tile x 6 N tiles (96 cols = one of q/k/v).
// Epilogue scatters into q (pre-scaled), k (row-major) and vT (transposed).
__global__ __launch_bounds__(256)
void k_qkv_gemm(const _Float16* __restrict__ h1, const _Float16* __restrict__ wT,
                const float* __restrict__ qkv_b, _Float16* __restrict__ qb,
                _Float16* __restrict__ kb, _Float16* __restrict__ vTb) {
  const int wave = blockIdx.x * 8 + (threadIdx.x >> 5);
  const int lane = threadIdx.x & 31;
  const int ngrp = wave % 3;        // 0:q 1:k 2:v
  const int m0   = (wave / 3) * 16;

  v8f acc[6];
#pragma unroll
  for (int i = 0; i < 6; ++i) acc[i] = vzero8();

#pragma unroll
  for (int kk = 0; kk < 3; ++kk) {
    v16h a = load_a_frag(h1, CDIM, m0, kk * 32, lane);
#pragma unroll
    for (int nt = 0; nt < 6; ++nt) {
      v16h b = load_b_frag(wT, CDIM, (ngrp * 6 + nt) * 16, kk * 32, lane);
      acc[nt] = wmma_f16(a, b, acc[nt]);
    }
  }

  const float scale = 0.17677669529663687f;  // (C/HEADS)^-0.5
  const int g = lane >> 4, c0 = lane & 15;
#pragma unroll
  for (int nt = 0; nt < 6; ++nt) {
    int rem  = nt * 16 + c0;          // 0..95 within this qkv third
    int head = rem >> 5, d = rem & 31;
    float bv = qkv_b[ngrp * CDIM + rem];
#pragma unroll
    for (int r = 0; r < 8; ++r) {
      int m = m0 + r + 8 * g;
      int w = m >> 6, t = m & 63;
      size_t base = (size_t)(w * 3 + head);
      float v = acc[nt][r] + bv;
      if (ngrp == 0)      qb [(base * 64 + t) * 32 + d] = (_Float16)(v * scale);
      else if (ngrp == 1) kb [(base * 64 + t) * 32 + d] = (_Float16)v;
      else                vTb[(base * 32 + d) * 64 + t] = (_Float16)v;  // Vt
    }
  }
}

// ---------------- windowed attention (one wave per window*head) ------------
#define ATT_WAVES 4
__global__ __launch_bounds__(ATT_WAVES * 32)
void k_attention(const _Float16* __restrict__ qb, const _Float16* __restrict__ kb,
                 const _Float16* __restrict__ vTb,
                 const float* __restrict__ biasHNN,
                 _Float16* __restrict__ attn_out) {
  __shared__ _Float16 Pbuf[ATT_WAVES][NTOK * NTOK];   // 4 x 8 KiB
  const int wid  = threadIdx.x >> 5;
  const int lane = threadIdx.x & 31;
  const int unit = blockIdx.x * ATT_WAVES + wid;      // window*3 + head
  const int w = unit / 3, head = unit % 3;

  const _Float16* q  = qb  + (size_t)unit * (NTOK * 32);
  const _Float16* k_ = kb  + (size_t)unit * (NTOK * 32);
  const _Float16* vT = vTb + (size_t)unit * (32 * NTOK);
  const float* bias  = biasHNN + head * (NTOK * NTOK);

  const int wimg = w & 255;
  const int wr = wimg >> 4, wc = wimg & 15;

  // ---- S = (q*scale) @ k^T : 4x4 tiles of 16x16, K=32 (one WMMA each) ----
  v8f sA[4][4];
#pragma unroll
  for (int mi = 0; mi < 4; ++mi)
#pragma unroll
    for (int ni = 0; ni < 4; ++ni) sA[mi][ni] = vzero8();

#pragma unroll
  for (int mi = 0; mi < 4; ++mi) {
    v16h a = load_a_frag(q, 32, mi * 16, 0, lane);
#pragma unroll
    for (int ni = 0; ni < 4; ++ni) {
      v16h b = load_b_frag(k_, 32, ni * 16, 0, lane);  // BT == k row-major
      sA[mi][ni] = wmma_f16(a, b, sA[mi][ni]);
    }
  }

  const int g = lane >> 4, c0 = lane & 15;
  // shifted-window group id (mask): region(c) = c<120?0 : c<124?1 : 2
#pragma unroll
  for (int mi = 0; mi < 4; ++mi) {
#pragma unroll
    for (int r = 0; r < 8; ++r) {
      int i  = mi * 16 + r + 8 * g;
      int ih = wr * WSZ + (i >> 3), iw = wc * WSZ + (i & 7);
      int gi = (ih < 120 ? 0 : (ih < 124 ? 1 : 2)) * 3 +
               (iw < 120 ? 0 : (iw < 124 ? 1 : 2));
      float vals[4];
      float rowmax = -1e30f;
#pragma unroll
      for (int ni = 0; ni < 4; ++ni) {
        int j  = ni * 16 + c0;
        int jh = wr * WSZ + (j >> 3), jw = wc * WSZ + (j & 7);
        int gj = (jh < 120 ? 0 : (jh < 124 ? 1 : 2)) * 3 +
                 (jw < 120 ? 0 : (jw < 124 ? 1 : 2));
        float v = sA[mi][ni][r] + bias[i * NTOK + j] + (gi != gj ? -100.f : 0.f);
        vals[ni] = v;
        rowmax = fmaxf(rowmax, v);
      }
      // row lives on one 16-lane half-wave: xor-reduce over masks 1,2,4,8
#pragma unroll
      for (int msk = 1; msk < 16; msk <<= 1)
        rowmax = fmaxf(rowmax, __shfl_xor(rowmax, msk, 32));
      float sum = 0.f;
#pragma unroll
      for (int ni = 0; ni < 4; ++ni) { vals[ni] = __expf(vals[ni] - rowmax); sum += vals[ni]; }
#pragma unroll
      for (int msk = 1; msk < 16; msk <<= 1) sum += __shfl_xor(sum, msk, 32);
      float inv = 1.f / sum;
#pragma unroll
      for (int ni = 0; ni < 4; ++ni)
        Pbuf[wid][i * NTOK + ni * 16 + c0] = (_Float16)(vals[ni] * inv);
    }
  }
  __syncthreads();

  // ---- out = P @ v : 4x2 tiles, K=64 (2 WMMA k-steps), B from vT ---------
  v8f o[4][2];
#pragma unroll
  for (int mi = 0; mi < 4; ++mi) { o[mi][0] = vzero8(); o[mi][1] = vzero8(); }

#pragma unroll
  for (int kk = 0; kk < 2; ++kk) {
    v16h b0 = load_b_frag(vT, NTOK, 0,  kk * 32, lane);
    v16h b1 = load_b_frag(vT, NTOK, 16, kk * 32, lane);
#pragma unroll
    for (int mi = 0; mi < 4; ++mi) {
      v16h a = load_a_frag(&Pbuf[wid][0], NTOK, mi * 16, kk * 32, lane);
      o[mi][0] = wmma_f16(a, b0, o[mi][0]);
      o[mi][1] = wmma_f16(a, b1, o[mi][1]);
    }
  }

#pragma unroll
  for (int mi = 0; mi < 4; ++mi)
#pragma unroll
    for (int ni = 0; ni < 2; ++ni)
#pragma unroll
      for (int r = 0; r < 8; ++r) {
        int i = mi * 16 + r + 8 * g;
        int d = ni * 16 + c0;
        attn_out[(size_t)(w * NTOK + i) * CDIM + head * 32 + d] =
            (_Float16)o[mi][ni][r];
      }
}

// ---------------- proj GEMM + window reverse + un-shift + residual ---------
__global__ __launch_bounds__(256)
void k_proj_gemm(const _Float16* __restrict__ attn,
                 const _Float16* __restrict__ wT,
                 const float* __restrict__ x, float* __restrict__ x1) {
  const int wave = blockIdx.x * 8 + (threadIdx.x >> 5);
  const int lane = threadIdx.x & 31;
  const int m0 = wave * 16;

  v8f acc[6];
#pragma unroll
  for (int i = 0; i < 6; ++i) acc[i] = vzero8();

#pragma unroll
  for (int kk = 0; kk < 3; ++kk) {
    v16h a = load_a_frag(attn, CDIM, m0, kk * 32, lane);
#pragma unroll
    for (int nt = 0; nt < 6; ++nt) {
      v16h b = load_b_frag(wT, CDIM, nt * 16, kk * 32, lane);
      acc[nt] = wmma_f16(a, b, acc[nt]);
    }
  }

  const int g = lane >> 4, c0 = lane & 15;
#pragma unroll
  for (int r = 0; r < 8; ++r) {
    int m = m0 + r + 8 * g;
    int wg = m >> 6, t = m & 63;
    int b = wg >> 8, wimg = wg & 255;
    int wr = wimg >> 4, wc = wimg & 15;
    int hh = wr * WSZ + (t >> 3), ww = wc * WSZ + (t & 7);
    int h0 = (hh + SHIFT_) & (HIMG - 1);
    int w0 = (ww + SHIFT_) & (HIMG - 1);
    size_t pix = ((size_t)b * (HIMG * HIMG) + h0 * HIMG + w0) * CDIM;
#pragma unroll
    for (int nt = 0; nt < 6; ++nt) {
      int col = nt * 16 + c0;
      x1[pix + col] = x[pix + col] + acc[nt][r];
    }
  }
}

// ---------------- fused MLP: h2@W1 -> GELU (LDS) -> @W2 + residual ---------
#define MLP_WAVES 4
__global__ __launch_bounds__(MLP_WAVES * 32)
void k_mlp_fused(const _Float16* __restrict__ h2, const _Float16* __restrict__ w1T,
                 const _Float16* __restrict__ w2T, float* __restrict__ xout) {
  __shared__ _Float16 mid[MLP_WAVES][16 * 384];   // 4 x 12 KiB GELU tile
  const int wid  = threadIdx.x >> 5;
  const int lane = threadIdx.x & 31;
  const int m0 = (blockIdx.x * MLP_WAVES + wid) * 16;
  const int g = lane >> 4, c0 = lane & 15;

  // phase 1: mid(16x384) = GELU(h2(16x96) @ W1)
  v16h af[3];
#pragma unroll
  for (int kk = 0; kk < 3; ++kk) af[kk] = load_a_frag(h2, CDIM, m0, kk * 32, lane);

  for (int nt = 0; nt < 24; ++nt) {
    v8f acc = vzero8();
#pragma unroll
    for (int kk = 0; kk < 3; ++kk) {
      v16h b = load_b_frag(w1T, CDIM, nt * 16, kk * 32, lane);
      acc = wmma_f16(af[kk], b, acc);
    }
#pragma unroll
    for (int r = 0; r < 8; ++r) {
      float v = acc[r];
      float gel = 0.5f * v *
          (1.f + tanhf(0.7978845608028654f * (v + 0.044715f * v * v * v)));
      mid[wid][(r + 8 * g) * 384 + nt * 16 + c0] = (_Float16)gel;
    }
  }
  __syncthreads();

  // phase 2: out(16x96) = mid @ W2, K=384 (12 k-steps), accumulate into x1
  v8f out[6];
#pragma unroll
  for (int i = 0; i < 6; ++i) out[i] = vzero8();

  for (int kk = 0; kk < 12; ++kk) {
    v16h a = load_a_frag(&mid[wid][0], 384, 0, kk * 32, lane);
#pragma unroll
    for (int nt = 0; nt < 6; ++nt) {
      v16h b = load_b_frag(w2T, 384, nt * 16, kk * 32, lane);
      out[nt] = wmma_f16(a, b, out[nt]);
    }
  }

#pragma unroll
  for (int nt = 0; nt < 6; ++nt)
#pragma unroll
    for (int r = 0; r < 8; ++r) {
      size_t idx = (size_t)(m0 + r + 8 * g) * CDIM + nt * 16 + c0;
      xout[idx] += out[nt][r];     // xout already holds x1 residual
    }
}

// ---------------------------------------------------------------------------

extern "C" void kernel_launch(void* const* d_in, const int* in_sizes, int n_in,
                              void* d_out, int out_size, void* d_ws, size_t ws_size,
                              hipStream_t stream) {
  const float* x          = (const float*)d_in[0];
  const float* ln1_g      = (const float*)d_in[1];
  const float* ln1_b      = (const float*)d_in[2];
  const float* qkv_w      = (const float*)d_in[3];
  const float* qkv_b      = (const float*)d_in[4];
  const float* proj_w     = (const float*)d_in[5];
  const float* bias_table = (const float*)d_in[6];
  const float* ln2_g      = (const float*)d_in[7];
  const float* ln2_b      = (const float*)d_in[8];
  const float* mlp_w1     = (const float*)d_in[9];
  const float* mlp_w2     = (const float*)d_in[10];
  float* xout = (float*)d_out;

  char* ws = (char*)d_ws;
  const size_t S16 = (size_t)MROWS * CDIM * 2;   // 48 MiB f16 activation tensor
  _Float16* h1  = (_Float16*)(ws + 0 * S16);     // LN1 windowed; reused as attn_out
  _Float16* qb  = (_Float16*)(ws + 1 * S16);     // q (scaled); reused as h2
  _Float16* kb  = (_Float16*)(ws + 2 * S16);
  _Float16* vTb = (_Float16*)(ws + 3 * S16);     // v transposed per head (32x64)
  _Float16* attn_out = h1;
  _Float16* h2       = qb;
  char* sm = ws + 4 * S16;
  _Float16* qkvwT  = (_Float16*)sm;  sm += (size_t)96 * 288 * 2;
  _Float16* projwT = (_Float16*)sm;  sm += (size_t)96 * 96 * 2;
  _Float16* w1T    = (_Float16*)sm;  sm += (size_t)96 * 384 * 2;
  _Float16* w2T    = (_Float16*)sm;  sm += (size_t)384 * 96 * 2;
  float*    biasHNN= (float*)sm;

  // 1) weight re-layout (f32 KxN -> f16 NxK) + bias gather
  k_weight_to_bt<<<(96 * 288 + 255) / 256, 256, 0, stream>>>(qkv_w,  qkvwT,  96, 288);
  k_weight_to_bt<<<(96 * 96  + 255) / 256, 256, 0, stream>>>(proj_w, projwT, 96, 96);
  k_weight_to_bt<<<(96 * 384 + 255) / 256, 256, 0, stream>>>(mlp_w1, w1T,    96, 384);
  k_weight_to_bt<<<(384 * 96 + 255) / 256, 256, 0, stream>>>(mlp_w2, w2T,    384, 96);
  k_bias_gather<<<(NHEAD * NTOK * NTOK + 255) / 256, 256, 0, stream>>>(bias_table, biasHNN);

  // 2) LN1 + cyclic shift + window partition -> h1 (f16, window-token major)
  k_layernorm<<<MROWS / 8, 256, 0, stream>>>(x, ln1_g, ln1_b, h1, 1);

  // 3) QKV GEMM (WMMA) -> q / k / vT
  k_qkv_gemm<<<(MROWS / 16) * 3 / 8, 256, 0, stream>>>(h1, qkvwT, qkv_b, qb, kb, vTb);

  // 4) windowed attention (WMMA qk^T, softmax, WMMA Pv) -> attn_out
  k_attention<<<(BATCH * NWIN * NHEAD) / ATT_WAVES, ATT_WAVES * 32, 0, stream>>>(
      qb, kb, vTb, biasHNN, attn_out);

  // 5) proj GEMM + window reverse + un-shift + residual -> d_out (= x1)
  k_proj_gemm<<<(MROWS / 16) / 8, 256, 0, stream>>>(attn_out, projwT, x, xout);

  // 6) LN2 -> h2 (f16)
  k_layernorm<<<MROWS / 8, 256, 0, stream>>>(xout, ln2_g, ln2_b, h2, 0);

  // 7) fused MLP (GEMM -> GELU in LDS -> GEMM) + residual -> d_out
  k_mlp_fused<<<(MROWS / 16) / MLP_WAVES, MLP_WAVES * 32, 0, stream>>>(
      h2, w1T, w2T, xout);
}
